// BasicBlock_73469710565660
// MI455X (gfx1250) — matrix operand order repaired and checked by reading, not compiled, not run on hardware
//
#include <hip/hip_runtime.h>

#define Bb 2
#define Cc 128
#define Nn 16384
#define Kk 12
#define TAPS 9
#define EPSc 1e-5f

typedef __attribute__((ext_vector_type(16))) __bf16 bf16x16;
typedef __attribute__((ext_vector_type(8)))  float  f32x8;
typedef __attribute__((ext_vector_type(4)))  float  f32x4;
typedef __attribute__((ext_vector_type(16))) short  s16x16;
typedef __attribute__((ext_vector_type(8)))  short  s16x8;

static __device__ __forceinline__ unsigned short f2bf(float f){
  union { float f; unsigned u; } v; v.f = f;
  unsigned r = v.u + 0x7FFFu + ((v.u >> 16) & 1u);   // round-to-nearest-even
  return (unsigned short)(r >> 16);
}
static __device__ __forceinline__ float bf2f(unsigned short h){
  union { unsigned u; float f; } v; v.u = ((unsigned)h) << 16;
  return v.f;
}
static __device__ __forceinline__ f32x8 wmma_bf16(bf16x16 a, bf16x16 b, f32x8 c){
  return __builtin_amdgcn_wmma_f32_16x16x32_bf16(false, a, false, b, (short)0, c, false, false);
}
// B-fragment (32x16 bf16, K x N): lane<16 -> col=lane, K=0..15 ; lane>=16 -> col=lane-16, K=16..31
// 16 contiguous bf16 per lane, fetched as two 16B LDS loads.
static __device__ __forceinline__ bf16x16 lds_bfrag(const unsigned short* p){
  s16x8 lo = *(const s16x8*)p;
  s16x8 hi = *(const s16x8*)(p + 8);
  s16x16 v = __builtin_shufflevector(lo, hi, 0,1,2,3,4,5,6,7,8,9,10,11,12,13,14,15);
  return __builtin_bit_cast(bf16x16, v);
}

// ---------------- weight pre-swizzle into contiguous-k A-fragment order (k = 16*(lane/16) + j) -----------------
// A (16x32 bf16, M x K): lane<16: M=lane, K = j ; lane>=16: M=lane-16, K = j+16
static __device__ __forceinline__ int a_kloc(int lane, int j){
  return (lane < 16) ? j : j + 16;
}

__global__ void k_pack_wg(const float* __restrict__ wg, unsigned short* __restrict__ out){
  int x = blockIdx.x * 256 + threadIdx.x;           // 128*256 = 32768 elems
  int j = x & 15, lane = (x >> 4) & 31, kk = (x >> 9) & 7, s = x >> 12;
  int c = kk * 32 + a_kloc(lane, j);
  int o = s * 16 + (lane & 15);
  out[x] = f2bf(wg[o * 256 + c]);
}

__global__ void k_pack_wc(const float* __restrict__ wc, unsigned short* __restrict__ out){
  int x = blockIdx.x * 256 + threadIdx.x;           // 128*1152 = 147456 elems
  int j = x & 15, lane = (x >> 4) & 31;
  int hi = x >> 9;                                  // s*36 + kk
  int kk = hi % 36, s = hi / 36;
  int K = kk * 32 + a_kloc(lane, j);
  int c = K & 127, k = K >> 7;                      // inner order K = k*128 + c
  int o = s * 16 + (lane & 15);
  out[x] = f2bf(wc[(o * Cc + c) * TAPS + k]);
}

// ---------------- transpose x [b,c,n] f32  ->  xT [b,n,c] bf16 --------------------
__global__ void k_transpose_x(const float* __restrict__ x, unsigned short* __restrict__ xT){
  __shared__ float t[32 * 9];
  int b  = blockIdx.x >> 9;                         // grid = B * N/32
  int n0 = (blockIdx.x & 511) * 32;
  int tid = threadIdx.x;
  for (int cb = 0; cb < 16; ++cb){
    int cc = tid >> 5, nl = tid & 31;
    t[nl * 9 + cc] = x[((size_t)(b * Cc + cb * 8 + cc)) * Nn + n0 + nl];
    __syncthreads();
    int nl2 = tid >> 3, cj = tid & 7;
    xT[((size_t)(b * Nn + n0 + nl2)) * Cc + cb * 8 + cj] = f2bf(t[nl2 * 9 + cj]);
    __syncthreads();
  }
}

__global__ void k_zero(float* p, int n){
  int i = blockIdx.x * 256 + threadIdx.x;
  if (i < n) p[i] = 0.f;
}

// ---------------- EdgeConv GEMM: 8 n per block, 96 cols (n,k), inner = 256 --------
#define E_COLS 96
#define E_BSTR 264      // 256 bf16 + 8 pad  -> conflict-free b128 fragment loads
__launch_bounds__(256, 1)
__global__ void k_edgeconv(const unsigned short* __restrict__ xT,
                           const int*   __restrict__ eidx,
                           const unsigned short* __restrict__ Apack,
                           float* __restrict__ hmax, float* __restrict__ hmin,
                           float* __restrict__ gsum, float* __restrict__ gssq)
{
  __shared__ __align__(16) unsigned char smem[E_COLS * E_BSTR * 2]; // 50688B; reused for D (96*132*4 = 50688B)
  __shared__ int   iid[E_COLS * 2];
  __shared__ float sSum[Cc], sSsq[Cc];
  unsigned short* ldsB = (unsigned short*)smem;
  float*          ldsD = (float*)smem;

  int tid = threadIdx.x;
  int b  = blockIdx.x >> 11;                        // grid = B * N/8
  int n0 = (blockIdx.x & 2047) * 8;

  if (tid < Cc){ sSum[tid] = 0.f; sSsq[tid] = 0.f; }
  if (tid < E_COLS){
    int nl = tid / 12, k = tid % 12;
    int base = (b * Nn + (n0 + nl)) * Kk + k;       // edge_index [2,B,N,K]
    iid[tid * 2 + 0] = eidx[Bb * Nn * Kk + base];   // e=1: center i
    iid[tid * 2 + 1] = eidx[base];                  // e=0: neighbor j
  }
  __syncthreads();

  // build B tile: col c<128 = xi ; c>=128 = xj - xi  (bf16)
  for (int task = tid; task < E_COLS * 16; task += 256){
    int col = task >> 4, seg = task & 15;
    int i1 = iid[col * 2], i0 = iid[col * 2 + 1];
    if (seg < 8){
      int c = seg * 8;
      *(s16x8*)(ldsB + col * E_BSTR + c) =
          *(const s16x8*)(xT + ((size_t)(b * Nn + i1)) * Cc + c);
    } else {
      int c = (seg - 8) * 8;
      s16x8 vi = *(const s16x8*)(xT + ((size_t)(b * Nn + i1)) * Cc + c);
      s16x8 vj = *(const s16x8*)(xT + ((size_t)(b * Nn + i0)) * Cc + c);
      s16x8 vo;
      #pragma unroll
      for (int q = 0; q < 8; ++q)
        vo[q] = (short)f2bf(bf2f((unsigned short)vj[q]) - bf2f((unsigned short)vi[q]));
      *(s16x8*)(ldsB + col * E_BSTR + 128 + c) = vo;
    }
  }
  __syncthreads();

  // GEMM: wave = o-strip, 6 col-tiles, K=256 (8 bf16 WMMA steps)
  int wave = tid >> 5, lane = tid & 31, s = wave;
  bf16x16 Afr[8];
  #pragma unroll
  for (int kk = 0; kk < 8; ++kk)
    Afr[kk] = __builtin_bit_cast(bf16x16,
        *(const s16x16*)(Apack + (((s * 8 + kk) * 32 + lane) << 4)));

  f32x8 acc[6];
  #pragma unroll
  for (int t = 0; t < 6; ++t) acc[t] = (f32x8)0.f;
  int lcol = lane & 15, chalf = (lane < 16) ? 0 : 16;
  #pragma unroll
  for (int t = 0; t < 6; ++t){
    const unsigned short* bbase = ldsB + (t * 16 + lcol) * E_BSTR + chalf;
    #pragma unroll
    for (int kk = 0; kk < 8; ++kk)
      acc[t] = wmma_bf16(Afr[kk], lds_bfrag(bbase + kk * 32), acc[t]);
  }
  __syncthreads();                                   // everyone done reading ldsB

  // stage D to LDS [col][o], padded stride 132 (8 contiguous floats per lane)
  int obase = s * 16 + ((lane < 16) ? 0 : 8);
  #pragma unroll
  for (int t = 0; t < 6; ++t){
    float* dp = ldsD + (t * 16 + lcol) * 132 + obase;
    *(f32x4*)dp       = __builtin_shufflevector(acc[t], acc[t], 0,1,2,3);
    *(f32x4*)(dp + 4) = __builtin_shufflevector(acc[t], acc[t], 4,5,6,7);
  }
  __syncthreads();

  // reduce: max/min over k per (n,o); accumulate BN stats (pre-relu, pre-max)
  int o = tid & 127, nlb = tid >> 7;
  float ls = 0.f, lq = 0.f;
  for (int it = 0; it < 4; ++it){
    int nl = nlb * 4 + it;
    float mx = -3.4e38f, mn = 3.4e38f;
    #pragma unroll
    for (int k = 0; k < 12; ++k){
      float v = ldsD[(nl * 12 + k) * 132 + o];
      mx = fmaxf(mx, v); mn = fminf(mn, v);
      ls += v; lq += v * v;
    }
    size_t oi = ((size_t)(b * Nn + n0 + nl)) * Cc + o;
    hmax[oi] = mx; hmin[oi] = mn;
  }
  atomicAdd(&sSum[o], ls);
  atomicAdd(&sSsq[o], lq);
  __syncthreads();
  if (tid < Cc){ atomicAdd(&gsum[tid], sSum[tid]); atomicAdd(&gssq[tid], sSsq[tid]); }
}

// ---------------- Weighted conv1d GEMM: 16 l per block, inner = 9*128 = 1152 ------
#define W_COLS 16
#define W_BSTR 1160     // 1152 bf16 + 8 pad
__launch_bounds__(256, 1)
__global__ void k_wconv(const unsigned short* __restrict__ yT,
                        const float* __restrict__ coords,
                        const unsigned short* __restrict__ Apack,
                        float* __restrict__ z,
                        float* __restrict__ gsum, float* __restrict__ gssq)
{
  __shared__ __align__(16) unsigned char smem[W_COLS * W_BSTR * 2]; // 37120B; reused for D (16*132*4)
  __shared__ float wts[W_COLS * TAPS];
  __shared__ float cl[W_COLS * 3];
  __shared__ float sSum[Cc], sSsq[Cc];
  unsigned short* ldsB = (unsigned short*)smem;
  float*          ldsD = (float*)smem;

  int tid = threadIdx.x;
  int b  = blockIdx.x >> 10;                        // grid = B * N/16
  int n0 = (blockIdx.x & 1023) * 16;

  if (tid < Cc){ sSum[tid] = 0.f; sSsq[tid] = 0.f; }
  if (tid < W_COLS * 3){
    int l = tid / 3, d = tid % 3;
    cl[l * 3 + d] = coords[((size_t)b * 3 + d) * Nn + n0 + l];
  }
  __syncthreads();
  // Gaussian tap weights (zero-padded coords participate in normalization, as in ref)
  for (int task = tid; task < W_COLS * TAPS; task += 256){
    int l = task / TAPS, k = task % TAPS;
    int lg = n0 + l + k - 4;
    float d2 = 0.f;
    if (lg >= 0 && lg < Nn){
      #pragma unroll
      for (int d = 0; d < 3; ++d){
        float dd = coords[((size_t)b * 3 + d) * Nn + lg] - cl[l * 3 + d];
        d2 += dd * dd;
      }
    } else {
      #pragma unroll
      for (int d = 0; d < 3; ++d){ float dd = cl[l * 3 + d]; d2 += dd * dd; }
    }
    wts[task] = __expf(-0.5f * d2);                 // SIGMA = 1
  }
  __syncthreads();
  if (tid < W_COLS){
    float sum = 0.f;
    #pragma unroll
    for (int k = 0; k < TAPS; ++k) sum += wts[tid * TAPS + k];
    float inv = 1.f / sum;
    #pragma unroll
    for (int k = 0; k < TAPS; ++k) wts[tid * TAPS + k] *= inv;
  }
  __syncthreads();

  // build B tile: inner index K = k*128 + c ; value = yT[l+k-4][c] * w[l][k] (0 at pads)
  for (int task = tid; task < W_COLS * TAPS * 16; task += 256){
    int col = task / (TAPS * 16);
    int r   = task - col * (TAPS * 16);
    int k = r >> 4, cs = r & 15;
    int c0 = cs * 8;
    int lg = n0 + col + k - 4;
    s16x8 vo;
    if (lg >= 0 && lg < Nn){
      float wv = wts[col * TAPS + k];
      s16x8 vy = *(const s16x8*)(yT + ((size_t)(b * Nn + lg)) * Cc + c0);
      #pragma unroll
      for (int q = 0; q < 8; ++q)
        vo[q] = (short)f2bf(bf2f((unsigned short)vy[q]) * wv);
    } else {
      #pragma unroll
      for (int q = 0; q < 8; ++q) vo[q] = 0;
    }
    *(s16x8*)(ldsB + col * W_BSTR + k * Cc + c0) = vo;
  }
  __syncthreads();

  // GEMM: 36 bf16 WMMA steps per wave
  int wave = tid >> 5, lane = tid & 31, s = wave;
  int lcol = lane & 15, chalf = (lane < 16) ? 0 : 16;
  const unsigned short* bbase = ldsB + lcol * W_BSTR + chalf;
  f32x8 acc = (f32x8)0.f;
  for (int kk = 0; kk < 36; ++kk){
    bf16x16 Af = __builtin_bit_cast(bf16x16,
        *(const s16x16*)(Apack + (((s * 36 + kk) * 32 + lane) << 4)));
    acc = wmma_bf16(Af, lds_bfrag(bbase + kk * 32), acc);
  }
  __syncthreads();
  int obase = s * 16 + ((lane < 16) ? 0 : 8);
  {
    float* dp = ldsD + lcol * 132 + obase;
    *(f32x4*)dp       = __builtin_shufflevector(acc, acc, 0,1,2,3);
    *(f32x4*)(dp + 4) = __builtin_shufflevector(acc, acc, 4,5,6,7);
  }
  __syncthreads();

  // store z [b][n][o] + BN statistics (conv bias is cancelled by BN; skipped)
  int o = tid & 127, cb = tid >> 7;
  float ls = 0.f, lq = 0.f;
  for (int it = 0; it < 8; ++it){
    int col = cb * 8 + it;
    float v = ldsD[col * 132 + o];
    z[((size_t)(b * Nn + n0 + col)) * Cc + o] = v;
    ls += v; lq += v * v;
  }
  atomicAdd(&sSum[o], ls); atomicAdd(&sSsq[o], lq);
  __syncthreads();
  if (tid < Cc){ atomicAdd(&gsum[tid], sSum[tid]); atomicAdd(&gssq[tid], sSsq[tid]); }
}

// ---------------- BN finalize / apply kernels -------------------------------------
__global__ void k_finalize(const float* __restrict__ gsum, const float* __restrict__ gssq,
                           const float* __restrict__ gamma, const float* __restrict__ beta,
                           float inv_cnt, float* __restrict__ coef){
  int o = threadIdx.x;                               // 128 threads
  float m = gsum[o] * inv_cnt;
  float v = gssq[o] * inv_cnt - m * m;
  float a = gamma[o] * rsqrtf(v + EPSc);
  coef[o] = a;
  coef[Cc + o] = beta[o] - a * m;
}

// max_k relu(a*h+b) == relu(a*(a>=0 ? hmax : hmin) + b)
__global__ void k_apply_edge(const float* __restrict__ hmax, const float* __restrict__ hmin,
                             const float* __restrict__ coef, unsigned short* __restrict__ yT){
  int i = blockIdx.x * 256 + threadIdx.x;
  int o = i & 127;
  float a = coef[o], bb = coef[Cc + o];
  float v = (a >= 0.f) ? hmax[i] : hmin[i];
  yT[i] = f2bf(fmaxf(fmaf(a, v, bb), 0.f));
}

__global__ void k_apply_bn_relu(const float* __restrict__ zin, const float* __restrict__ coef,
                                unsigned short* __restrict__ outT){
  int i = blockIdx.x * 256 + threadIdx.x;
  int o = i & 127;
  outT[i] = f2bf(fmaxf(fmaf(coef[o], zin[i], coef[Cc + o]), 0.f));
}

// out[b,o,n] = relu(a*z2 + b + x)  with LDS transpose so both sides are coalesced-ish
__global__ void k_final(const float* __restrict__ z2, const float* __restrict__ coef,
                        const float* __restrict__ x, float* __restrict__ out){
  __shared__ float Z[32 * 132];
  int tid = threadIdx.x;
  int b  = blockIdx.x >> 9;                         // grid = B * N/32
  int n0 = (blockIdx.x & 511) * 32;
  for (int it = 0; it < 16; ++it){
    int nl = (tid >> 7) + it * 2;
    int o  = tid & 127;
    Z[nl * 132 + o] = z2[((size_t)(b * Nn + n0 + nl)) * Cc + o];
  }
  __syncthreads();
  int o = tid >> 1, nh = (tid & 1) * 16;
  float a = coef[o], bb = coef[Cc + o];
  size_t xb = ((size_t)b * Cc + o) * Nn + n0;
  for (int j = 0; j < 16; ++j){
    int n = nh + j;
    float v = fmaf(a, Z[n * 132 + o], bb) + x[xb + n];
    out[xb + n] = fmaxf(v, 0.f);
  }
}

// ----------------------------------- host -----------------------------------------
extern "C" void kernel_launch(void* const* d_in, const int* in_sizes, int n_in,
                              void* d_out, int out_size, void* d_ws, size_t ws_size,
                              hipStream_t stream)
{
  const float* x      = (const float*)d_in[0];
  const float* coords = (const float*)d_in[1];
  const int*   eidx   = (const int*)  d_in[2];
  const float* wg1    = (const float*)d_in[3];
  const float* gg1    = (const float*)d_in[4];
  const float* bg1    = (const float*)d_in[5];
  const float* wc1    = (const float*)d_in[6];
  /* bc1 = d_in[7]  : cancelled by BN mean subtraction */
  const float* bn1g   = (const float*)d_in[8];
  const float* bn1b   = (const float*)d_in[9];
  const float* wg2    = (const float*)d_in[10];
  const float* gg2    = (const float*)d_in[11];
  const float* bg2    = (const float*)d_in[12];
  const float* wc2    = (const float*)d_in[13];
  /* bc2 = d_in[14] : cancelled by BN mean subtraction */
  const float* bn2g   = (const float*)d_in[15];
  const float* bn2b   = (const float*)d_in[16];
  float* out = (float*)d_out;
  (void)in_sizes; (void)n_in; (void)out_size; (void)ws_size;

  char* ws = (char*)d_ws;
  size_t off = 0;
  auto alloc = [&](size_t bytes){ char* p = ws + off; off += (bytes + 255) & ~(size_t)255; return p; };
  const size_t BNC = (size_t)Bb * Nn * Cc;

  float*          hmax  = (float*)         alloc(BNC * 4);
  float*          hmin  = (float*)         alloc(BNC * 4);
  float*          zbuf  = (float*)         alloc(BNC * 4);
  unsigned short* xT    = (unsigned short*)alloc(BNC * 2);
  unsigned short* actT  = (unsigned short*)alloc(BNC * 2);
  unsigned short* ybuf  = (unsigned short*)alloc(BNC * 2);
  unsigned short* Ag1   = (unsigned short*)alloc(32768u  * 2);
  unsigned short* Ag2   = (unsigned short*)alloc(32768u  * 2);
  unsigned short* Ac1   = (unsigned short*)alloc(147456u * 2);
  unsigned short* Ac2   = (unsigned short*)alloc(147456u * 2);
  float*          stats = (float*)         alloc(1024 * 4);   // 4 stages x (sum,ssq)[128]
  float*          coefs = (float*)         alloc(1024 * 4);   // 4 x (a,b)[128]
  float *s0s = stats +   0, *s0q = stats + 128;
  float *s1s = stats + 256, *s1q = stats + 384;
  float *s2s = stats + 512, *s2q = stats + 640;
  float *s3s = stats + 768, *s3q = stats + 896;
  float *c0 = coefs, *c1 = coefs + 256, *c2 = coefs + 512, *c3 = coefs + 768;

  const float invE  = 1.f / (float)((size_t)Bb * Nn * Kk);   // EdgeConv BN count
  const float invBN = 1.f / (float)((size_t)Bb * Nn);        // BN1d count

  k_zero       <<<4,    256, 0, stream>>>(stats, 1024);
  k_pack_wg    <<<128,  256, 0, stream>>>(wg1, Ag1);
  k_pack_wg    <<<128,  256, 0, stream>>>(wg2, Ag2);
  k_pack_wc    <<<576,  256, 0, stream>>>(wc1, Ac1);
  k_pack_wc    <<<576,  256, 0, stream>>>(wc2, Ac2);
  k_transpose_x<<<1024, 256, 0, stream>>>(x, xT);

  // stage 1
  k_edgeconv   <<<4096, 256, 0, stream>>>(xT, eidx, Ag1, hmax, hmin, s0s, s0q);
  k_finalize   <<<1,    128, 0, stream>>>(s0s, s0q, gg1, bg1, invE, c0);
  k_apply_edge <<<16384,256, 0, stream>>>(hmax, hmin, c0, ybuf);
  k_wconv      <<<2048, 256, 0, stream>>>(ybuf, coords, Ac1, zbuf, s1s, s1q);
  k_finalize   <<<1,    128, 0, stream>>>(s1s, s1q, bn1g, bn1b, invBN, c1);
  k_apply_bn_relu<<<16384,256,0, stream>>>(zbuf, c1, actT);

  // stage 2
  k_edgeconv   <<<4096, 256, 0, stream>>>(actT, eidx, Ag2, hmax, hmin, s2s, s2q);
  k_finalize   <<<1,    128, 0, stream>>>(s2s, s2q, gg2, bg2, invE, c2);
  k_apply_edge <<<16384,256, 0, stream>>>(hmax, hmin, c2, ybuf);
  k_wconv      <<<2048, 256, 0, stream>>>(ybuf, coords, Ac2, zbuf, s3s, s3q);
  k_finalize   <<<1,    128, 0, stream>>>(s3s, s3q, bn2g, bn2b, invBN, c3);
  k_final      <<<1024, 256, 0, stream>>>(zbuf, c3, x, out);

  // tuple outputs 2 & 3: pass-through coords and edge_index
  hipMemcpyAsync(out + (size_t)Bb * Cc * Nn, coords,
                 (size_t)Bb * 3 * Nn * 4, hipMemcpyDeviceToDevice, stream);
  hipMemcpyAsync(out + (size_t)Bb * Cc * Nn + (size_t)Bb * 3 * Nn, eidx,
                 (size_t)2 * Bb * Nn * Kk * 4, hipMemcpyDeviceToDevice, stream);
}
